// GNNBackbone_72018011619520
// MI455X (gfx1250) — compile-verified, matrix operand order checked by reading, not run on hardware
//
#include <hip/hip_runtime.h>
#include <hip/hip_bf16.h>

typedef __attribute__((ext_vector_type(16))) _Float16 v16h;
typedef __attribute__((ext_vector_type(8)))  _Float16 v8h;
typedef __attribute__((ext_vector_type(8)))  float    v8f;
typedef __attribute__((ext_vector_type(4)))  int      v4i;

#define MPAD        100096      // 782 * 128
#define HID         128
#define K_IN        224         // 205 padded to 7*32
#define NKT_IN      7
#define NKT_H       4
#define TILE_HALVES 512         // 32 lanes * 16 halves per B tile
#define LAYER_WP    16384       // 4 ktiles * 8 ntiles * 512 halves
#define IN_WP       (NKT_IN * 8 * TILE_HALVES)   // 28672 halves

#if defined(__has_builtin)
#  if __has_builtin(__builtin_amdgcn_global_load_async_to_lds_b128)
#    define ASYNC_LDS 1
#  endif
#  if __has_builtin(__builtin_amdgcn_s_wait_asynccnt)
#    define HAVE_WAIT_ASYNC 1
#  endif
#endif

typedef __attribute__((address_space(1))) v4i* gas_v4i_p;
typedef __attribute__((address_space(3))) v4i* las_v4i_p;

__device__ __forceinline__ float gelu_exact(float x) {
    return 0.5f * x * (1.0f + erff(x * 0.70710678118654752440f));
}

// Cooperative copy of packed weights (global f16) into LDS, 16B chunks / thread.
// Uses gfx1250 async global->LDS DMA when the builtin exists.
__device__ __forceinline__ void stage_weights(_Float16* dst /*LDS*/,
                                              const _Float16* __restrict__ src,
                                              int halves, int tid) {
#ifdef ASYNC_LDS
    for (int c = tid; c < (halves >> 3); c += 256) {
        __builtin_amdgcn_global_load_async_to_lds_b128(
            (gas_v4i_p)(src + (size_t)c * 8),
            (las_v4i_p)(dst + (size_t)c * 8),
            0, 0);
    }
#  ifdef HAVE_WAIT_ASYNC
    __builtin_amdgcn_s_wait_asynccnt(0);
#  else
    asm volatile("s_wait_asynccnt 0x0" ::: "memory");
#  endif
#else
    for (int c = tid; c < (halves >> 3); c += 256) {
        *(v8h*)(dst + (size_t)c * 8) = *(const v8h*)(src + (size_t)c * 8);
    }
#endif
    __syncthreads();
}

// ---------------- weight pack: W (K x 128 row-major f32) -> WMMA B-operand f16 tiles ---
__global__ void k_pack(const float* __restrict__ W, _Float16* __restrict__ out,
                       int Kvalid, int nElems) {
    int i = blockIdx.x * 256 + threadIdx.x;
    if (i >= nElems) return;
    int j    = i & 15;
    int lane = (i >> 4) & 31;
    int nt   = (i >> 9) & 7;
    int kt   = i >> 12;
    int K = kt * 32 + (lane >> 4) * 8 + (j >> 3) * 16 + (j & 7);
    int N = nt * 16 + (lane & 15);
    float v = (K < Kvalid) ? W[(size_t)K * HID + N] : 0.0f;
    out[i] = (_Float16)v;
}

// ---------------- padded f16 input rows [feat(140)|embed(64)|depth(1)|0 pad] -----------
// one block per row, thread = column -> fully coalesced
__global__ void k_prep(const float* __restrict__ feat, const float* __restrict__ topo,
                       const float* __restrict__ embed, const int* __restrict__ opc,
                       _Float16* __restrict__ H0, int n) {
    int row = blockIdx.x;
    int j   = threadIdx.x;          // 0..255, columns 0..223 valid
    if (j >= K_IN) return;
    float v = 0.0f;
    if (row < n) {
        if (j < 140) {
            v = feat[(size_t)row * 140 + j];
        } else if (j < 204) {
            int op = opc[row];
            op = op < 0 ? 0 : (op > 127 ? 127 : op);
            v = embed[(size_t)op * 64 + (j - 140)];
        } else if (j == 204) {
            v = topo[row];
        }
    }
    H0[(size_t)row * K_IN + j] = (_Float16)v;
}

__global__ void k_zero(float* __restrict__ p, size_t count) {
    size_t i = (size_t)blockIdx.x * 256 + threadIdx.x;
    if (i < count) p[i] = 0.0f;
}

__global__ void k_deg(const int* __restrict__ tgt, float* __restrict__ deg, int nE) {
    int e = blockIdx.x * 256 + threadIdx.x;
    if (e < nE) atomicAdd(&deg[tgt[e]], 1.0f);
}

__global__ void k_invdeg(float* __restrict__ deg, int n) {
    int i = blockIdx.x * 256 + threadIdx.x;
    if (i < n) deg[i] = 1.0f / fmaxf(deg[i], 1.0f);
}

// ---------------- edge scatter: agg[tgt] += h[src] (block = 8 edges, thread = feature) -
__global__ void k_scatter(const int* __restrict__ src, const int* __restrict__ tgt,
                          const float* __restrict__ hF, float* __restrict__ agg, int nE) {
    int e0 = blockIdx.x * 8;
    int t  = threadIdx.x;           // 0..127
    #pragma unroll
    for (int i = 0; i < 8; ++i) {
        int e = e0 + i;
        if (e >= nE) return;
        int s = src[e], d = tgt[e];
        atomicAdd(&agg[(size_t)d * HID + t], hF[(size_t)s * HID + t]);
    }
}

// ---------------- agg *= inv_deg; convert to f16 --------------------------------------
__global__ void k_scale(const float* __restrict__ agg, const float* __restrict__ inv,
                        _Float16* __restrict__ aggH, int n) {
    size_t i = (size_t)blockIdx.x * 256 + threadIdx.x;
    if (i >= (size_t)MPAD * HID) return;
    int row = (int)(i >> 7);
    float s = (row < n) ? inv[row] : 0.0f;
    aggH[i] = (_Float16)(agg[i] * s);
}

// ---------------- WMMA helpers --------------------------------------------------------
__device__ __forceinline__ v16h load_a(const _Float16* base) {
    v8h lo = *(const v8h*)base;
    v8h hi = *(const v8h*)(base + 16);
    v16h a;
    #pragma unroll
    for (int i = 0; i < 8; ++i) { a[i] = lo[i]; a[i + 8] = hi[i]; }
    return a;
}

__device__ __forceinline__ v16h lds_b(const _Float16* sB, int tile, int lane) {
    return *(const v16h*)(sB + (size_t)tile * TILE_HALVES + lane * 16);
}

// ---------------- input projection: H0(MPADxK_IN) @ WinP -> bias,LN,GELU -> hF,hH ------
__global__ __launch_bounds__(256) void k_gemm_in(
    const _Float16* __restrict__ H0, const _Float16* __restrict__ WinP,
    const float* __restrict__ bin, const float* __restrict__ g0,
    const float* __restrict__ be0, float* __restrict__ hF, _Float16* __restrict__ hH) {
    __shared__ __align__(32) _Float16 sW[IN_WP];          // 56 KB
    const int tid  = threadIdx.x;
    const int lane = tid & 31;
    const int w    = tid >> 5;
    const int m0   = blockIdx.x * 128 + w * 16;           // wave owns rows m0..m0+15
    const int rIn  = lane & 15;
    const int grp  = lane >> 4;

    stage_weights(sW, WinP, IN_WP, tid);

    v8f acc[8] = {};
    const _Float16* aBase = H0 + (size_t)(m0 + rIn) * K_IN + grp * 8;
    for (int kt = 0; kt < NKT_IN; ++kt) {
        v16h a = load_a(aBase + kt * 32);
        if (kt + 1 < NKT_IN) __builtin_prefetch(aBase + (kt + 1) * 32, 0, 0);
        #pragma unroll
        for (int nt = 0; nt < 8; ++nt) {
            v16h b = lds_b(sW, kt * 8 + nt, lane);
            acc[nt] = __builtin_amdgcn_wmma_f32_16x16x32_f16(
                false, a, false, b, (short)0, acc[nt], false, false);
        }
    }
    // epilogue: x = acc + b_in; LayerNorm; GELU
    #pragma unroll
    for (int r = 0; r < 8; ++r) {
        float s = 0.0f, s2 = 0.0f;
        #pragma unroll
        for (int nt = 0; nt < 8; ++nt) {
            int c = nt * 16 + rIn;
            float v = acc[nt][r] + bin[c];
            acc[nt][r] = v;
            s += v; s2 += v * v;
        }
        #pragma unroll
        for (int m = 1; m <= 8; m <<= 1) {
            s  += __shfl_xor(s,  m, 32);
            s2 += __shfl_xor(s2, m, 32);
        }
        float mu  = s  * (1.0f / HID);
        float var = s2 * (1.0f / HID) - mu * mu;
        float rsv = rsqrtf(var + 1e-5f);
        size_t base = (size_t)(m0 + 8 * grp + r) * HID;
        #pragma unroll
        for (int nt = 0; nt < 8; ++nt) {
            int c = nt * 16 + rIn;
            float y = (acc[nt][r] - mu) * rsv * g0[c] + be0[c];
            y = gelu_exact(y);
            hF[base + c] = y;
            hH[base + c] = (_Float16)y;
        }
    }
}

// ---------------- layer: out = aggH@Wl + hH@Wr + bl; h = LN(h + gelu(out)) -------------
__global__ __launch_bounds__(256) void k_gemm_layer(
    const _Float16* __restrict__ aggH, const _Float16* __restrict__ hHin,
    const _Float16* __restrict__ WlP, const _Float16* __restrict__ WrP,
    const float* __restrict__ bl, const float* __restrict__ g,
    const float* __restrict__ be, float* __restrict__ hF, _Float16* __restrict__ hH) {
    __shared__ __align__(32) _Float16 sW[2 * LAYER_WP];   // 64 KB: [Wl | Wr]
    const int tid  = threadIdx.x;
    const int lane = tid & 31;
    const int w    = tid >> 5;
    const int m0   = blockIdx.x * 128 + w * 16;
    const int rIn  = lane & 15;
    const int grp  = lane >> 4;

    stage_weights(sW, WlP, LAYER_WP, tid);
    stage_weights(sW + LAYER_WP, WrP, LAYER_WP, tid);

    v8f acc[8] = {};
    const _Float16* aA = aggH + (size_t)(m0 + rIn) * HID + grp * 8;
    const _Float16* aH = hHin + (size_t)(m0 + rIn) * HID + grp * 8;
    for (int kt = 0; kt < NKT_H; ++kt) {
        v16h a1 = load_a(aA + kt * 32);
        v16h a2 = load_a(aH + kt * 32);
        if (kt + 1 < NKT_H) {
            __builtin_prefetch(aA + (kt + 1) * 32, 0, 0);
            __builtin_prefetch(aH + (kt + 1) * 32, 0, 0);
        }
        #pragma unroll
        for (int nt = 0; nt < 8; ++nt) {
            v16h b1 = lds_b(sW, kt * 8 + nt, lane);
            v16h b2 = lds_b(sW + LAYER_WP, kt * 8 + nt, lane);
            acc[nt] = __builtin_amdgcn_wmma_f32_16x16x32_f16(
                false, a1, false, b1, (short)0, acc[nt], false, false);
            acc[nt] = __builtin_amdgcn_wmma_f32_16x16x32_f16(
                false, a2, false, b2, (short)0, acc[nt], false, false);
        }
    }
    #pragma unroll
    for (int r = 0; r < 8; ++r) {
        size_t base = (size_t)(m0 + 8 * grp + r) * HID;
        float s = 0.0f, s2 = 0.0f;
        #pragma unroll
        for (int nt = 0; nt < 8; ++nt) {
            int c = nt * 16 + rIn;
            float v = hF[base + c] + gelu_exact(acc[nt][r] + bl[c]);  // residual + GELU
            acc[nt][r] = v;
            s += v; s2 += v * v;
        }
        #pragma unroll
        for (int m = 1; m <= 8; m <<= 1) {
            s  += __shfl_xor(s,  m, 32);
            s2 += __shfl_xor(s2, m, 32);
        }
        float mu  = s  * (1.0f / HID);
        float var = s2 * (1.0f / HID) - mu * mu;
        float rsv = rsqrtf(var + 1e-5f);
        #pragma unroll
        for (int nt = 0; nt < 8; ++nt) {
            int c = nt * 16 + rIn;
            float y = (acc[nt][r] - mu) * rsv * g[c] + be[c];
            hF[base + c] = y;
            hH[base + c] = (_Float16)y;
        }
    }
}

__global__ void k_copy(const float* __restrict__ src, float* __restrict__ dst, size_t count) {
    size_t i = (size_t)blockIdx.x * 256 + threadIdx.x;
    if (i < count) dst[i] = src[i];
}

// ---------------- workspace offsets (256B aligned) ------------------------------------
#define OFF_WINP  ((size_t)0)                         // 7*8*512*2   = 57344
#define OFF_WLP   ((size_t)57344)                     // 4*16384*2   = 131072
#define OFF_WRP   ((size_t)188416)                    // 131072
#define OFF_DEG   ((size_t)319488)                    // n*4 -> 400384
#define OFF_H0    ((size_t)719872)                    // MPAD*224*2  = 44843008
#define OFF_HF    ((size_t)45562880)                  // MPAD*128*4  = 51249152
#define OFF_HH    ((size_t)96812032)                  // MPAD*128*2  = 25624576
#define OFF_AGGF  ((size_t)122436608)                 // 51249152
#define OFF_AGGH  ((size_t)173685760)                 // 25624576  (end ~199.3MB)

extern "C" void kernel_launch(void* const* d_in, const int* in_sizes, int n_in,
                              void* d_out, int out_size, void* d_ws, size_t ws_size,
                              hipStream_t stream) {
    const float* node_feat = (const float*)d_in[0];
    const float* topo      = (const float*)d_in[1];
    const float* embed     = (const float*)d_in[2];
    const float* W_in      = (const float*)d_in[3];
    const float* b_in      = (const float*)d_in[4];
    const float* ln0_g     = (const float*)d_in[5];
    const float* ln0_b     = (const float*)d_in[6];
    const float* Wl        = (const float*)d_in[7];
    const float* bl        = (const float*)d_in[8];
    const float* Wr        = (const float*)d_in[9];
    const float* ln_g      = (const float*)d_in[10];
    const float* ln_b      = (const float*)d_in[11];
    const int*   opc       = (const int*)d_in[12];
    const int*   edges     = (const int*)d_in[13];
    const int n  = in_sizes[12];
    const int nE = in_sizes[13] / 2;

    char* ws = (char*)d_ws;
    _Float16* WinP = (_Float16*)(ws + OFF_WINP);
    _Float16* WlP  = (_Float16*)(ws + OFF_WLP);
    _Float16* WrP  = (_Float16*)(ws + OFF_WRP);
    float*    deg  = (float*)(ws + OFF_DEG);
    _Float16* H0   = (_Float16*)(ws + OFF_H0);
    float*    hF   = (float*)(ws + OFF_HF);
    _Float16* hH   = (_Float16*)(ws + OFF_HH);
    float*    aggF = (float*)(ws + OFF_AGGF);
    _Float16* aggH = (_Float16*)(ws + OFF_AGGH);

    // pack weights into WMMA B-operand layout
    k_pack<<<(IN_WP + 255) / 256, 256, 0, stream>>>(W_in, WinP, 205, IN_WP);
    for (int i = 0; i < 4; ++i) {
        k_pack<<<(LAYER_WP + 255) / 256, 256, 0, stream>>>(
            Wl + (size_t)i * HID * HID, WlP + (size_t)i * LAYER_WP, HID, LAYER_WP);
        k_pack<<<(LAYER_WP + 255) / 256, 256, 0, stream>>>(
            Wr + (size_t)i * HID * HID, WrP + (size_t)i * LAYER_WP, HID, LAYER_WP);
    }

    // padded f16 input matrix + degrees
    k_prep<<<MPAD, 256, 0, stream>>>(node_feat, topo, embed, opc, H0, n);
    k_zero<<<(n + 255) / 256, 256, 0, stream>>>(deg, (size_t)n);
    k_deg<<<(nE + 255) / 256, 256, 0, stream>>>(edges + nE, deg, nE);
    k_invdeg<<<(n + 255) / 256, 256, 0, stream>>>(deg, n);

    // input projection (WMMA) + LN + GELU
    k_gemm_in<<<MPAD / 128, 256, 0, stream>>>(H0, WinP, b_in, ln0_g, ln0_b, hF, hH);

    // 4 SAGE layers
    const size_t aggCount = (size_t)MPAD * HID;
    for (int i = 0; i < 4; ++i) {
        k_zero<<<(unsigned)((aggCount + 255) / 256), 256, 0, stream>>>(aggF, aggCount);
        k_scatter<<<(nE + 7) / 8, 128, 0, stream>>>(edges, edges + nE, hF, aggF, nE);
        k_scale<<<(unsigned)((aggCount + 255) / 256), 256, 0, stream>>>(aggF, deg, aggH, n);
        k_gemm_layer<<<MPAD / 128, 256, 0, stream>>>(
            aggH, hH, WlP + (size_t)i * LAYER_WP, WrP + (size_t)i * LAYER_WP,
            bl + (size_t)i * HID, ln_g + (size_t)i * HID, ln_b + (size_t)i * HID, hF, hH);
    }

    k_copy<<<(unsigned)(((size_t)n * HID + 255) / 256), 256, 0, stream>>>(
        hF, (float*)d_out, (size_t)n * HID);
}
